// AdditiveAttention_17961553232521
// MI455X (gfx1250) — matrix operand order repaired and checked by reading, not compiled, not run on hardware
//
#include <hip/hip_runtime.h>
#include <hip/hip_bf16.h>

// ---------------------------------------------------------------------------
// Additive attention, MI455X (gfx1250, wave32, WMMA).
// Pipeline:
//   1) cvt_kernel     : value fp32 -> bf16 (for WMMA B operand)
//   2) proj_kernel x2 : q_ = query@Wq^T, k_ = key@Wk^T   (v_wmma_f32_16x16x32_bf16)
//   3) logits_kernel  : logits[b,q,k] = sum_a w[a]*tanh(q_+k_+bias)+wb
//                       (LDS-transposed k_ tiles, tanh-bound VALU stage)
//   4) softmax_kernel : row softmax, writes fp32 attn (output #2) + bf16 copy
//   5) av_kernel      : out = attn @ value                (v_wmma_f32_16x16x32_bf16)
// ---------------------------------------------------------------------------

#define B_   8
#define TQ_  512
#define TK_  512
#define DD_  512
#define AD_  128

typedef __attribute__((ext_vector_type(16))) __bf16 v16bf;
typedef __attribute__((ext_vector_type(8)))  __bf16 v8bf;
typedef __attribute__((ext_vector_type(4)))  __bf16 v4bf;
typedef __attribute__((ext_vector_type(2)))  __bf16 v2bf;
typedef __attribute__((ext_vector_type(8)))  float  v8f;

__device__ __forceinline__ float fast_tanh(float x) {
#if __has_builtin(__builtin_amdgcn_tanhf)
    return __builtin_amdgcn_tanhf(x);
#else
    // tanh(x) = 1 - 2/(exp2(2x*log2e)+1) : 2 trans ops (v_exp_f32, v_rcp_f32)
    float e = __builtin_amdgcn_exp2f(x * 2.8853900817779268f);
    return 1.0f - 2.0f * __builtin_amdgcn_rcpf(e + 1.0f);
#endif
}

// --------------------------- 1) value -> bf16 ------------------------------
__global__ __launch_bounds__(256) void cvt_kernel(const float* __restrict__ src,
                                                  __bf16* __restrict__ dst) {
    size_t i = (size_t)blockIdx.x * 256 + threadIdx.x;     // one float4 per thread
    float4 v = ((const float4*)src)[i];
    v4bf o;
    o[0] = (__bf16)v.x; o[1] = (__bf16)v.y; o[2] = (__bf16)v.z; o[3] = (__bf16)v.w;
    ((v4bf*)dst)[i] = o;
}

// ------------------- 2) projection GEMM: [4096x512]@[512x128] --------------
// Block: 256 thr (8 waves). Wave computes 16 rows x 128 cols (8 accum tiles).
// W^T chunk [32k x 128n] staged in LDS as bf16 each K-step.
__global__ __launch_bounds__(256) void proj_kernel(const float* __restrict__ X,
                                                   const float* __restrict__ W,
                                                   float* __restrict__ OUT) {
    __shared__ __bf16 WT[32][136];                         // pad row to 136 (16B mult)
    const int t = threadIdx.x, wave = t >> 5, lane = t & 31;
    const int m0    = blockIdx.x * 128 + wave * 16;
    const int mrow  = m0 + (lane & 15);
    const int kbase = (lane < 16) ? 0 : 8;
    const int hi8   = (lane < 16) ? 0 : 8;

    v8f acc[8];
#pragma unroll
    for (int i = 0; i < 8; ++i)
#pragma unroll
        for (int e = 0; e < 8; ++e) acc[i][e] = 0.0f;

    for (int kc = 0; kc < 16; ++kc) {
        const int k0 = kc * 32;
        __syncthreads();
        // stage W^T[k0..k0+31][0..127] = W[n][k0+k]
        for (int idx = t; idx < 32 * 128; idx += 256) {
            int kk = idx & 31, nn = idx >> 5;
            WT[kk][nn] = (__bf16)W[nn * DD_ + k0 + kk];
        }
        __syncthreads();

        // A fragment (16x32 bf16) per the ISA 16-bit A layout
        v16bf afr;
        const float* xr = X + (size_t)mrow * DD_ + k0 + kbase;
#pragma unroll
        for (int p = 0; p < 8; ++p) {
            int kp = (p < 4) ? 2 * p : 16 + 2 * (p - 4);
            float2 xv = *(const float2*)(xr + kp);
            afr[2 * p]     = (__bf16)xv.x;
            afr[2 * p + 1] = (__bf16)xv.y;
        }
        // B fragments: lane l holds N=l%16, K in the same table order as A
#pragma unroll
        for (int nb = 0; nb < 8; ++nb) {
            v16bf bfr;
#pragma unroll
            for (int s = 0; s < 16; ++s)
                bfr[s] = WT[kbase + ((s < 8) ? s : 8 + s)][nb * 16 + (lane & 15)];
            acc[nb] = __builtin_amdgcn_wmma_f32_16x16x32_bf16(
                false, afr, false, bfr, (short)0, acc[nb], false, false);
        }
    }
    // C layout: lane<16 -> N=lane, M=r ; lane>=16 -> N=lane-16, M=8+r
#pragma unroll
    for (int nb = 0; nb < 8; ++nb)
#pragma unroll
        for (int r = 0; r < 8; ++r)
            OUT[(size_t)(m0 + r + hi8) * AD_ + nb * 16 + (lane & 15)] = acc[nb][r];
}

// ------------------- 3) logits: tanh-bound VALU stage ----------------------
// Grid (64, 8): block handles 8 q-rows of one batch; 8 waves, 1 q-row/wave.
// k_ staged transposed [a][k] in LDS, 64 k at a time (8 passes).
#define QB_  8
#define KCH_ 64
__global__ __launch_bounds__(256) void logits_kernel(const float* __restrict__ qws,
                                                     const float* __restrict__ kws,
                                                     const float* __restrict__ bias,
                                                     const float* __restrict__ ww,
                                                     const float* __restrict__ wb,
                                                     float* __restrict__ logits) {
    __shared__ float  kT[AD_][KCH_ + 4];                   // 34.8 KB, pad for banks
    __shared__ float2 qw[QB_][AD_];                        // (q+bias, w) pairs, 8 KB
    const int t = threadIdx.x, wave = t >> 5, lane = t & 31;
    const int b = blockIdx.y, q0 = blockIdx.x * QB_;

    for (int idx = t; idx < QB_ * AD_; idx += 256) {
        int qi = idx >> 7, a = idx & 127;
        qw[qi][a] = make_float2(
            qws[((size_t)(b * TQ_ + q0 + qi)) * AD_ + a] + bias[a], ww[a]);
    }
    const float wbias = wb[0];
    const float2* qp = qw[wave];                           // this wave's q row

    for (int kh = 0; kh < TK_ / KCH_; ++kh) {
        __syncthreads();
        for (int idx = t; idx < KCH_ * AD_; idx += 256) {  // coalesced gather+transpose
            int k = idx >> 7, a = idx & 127;
            kT[a][k] = kws[((size_t)(b * TK_ + kh * KCH_ + k)) * AD_ + a];
        }
        __syncthreads();

        float a0 = 0.0f, a1 = 0.0f;                        // 2 k-values per lane
#pragma unroll 8
        for (int a = 0; a < AD_; ++a) {
            float2 kv = *(const float2*)&kT[a][2 * lane];  // conflict-free b64
            float2 qv = qp[a];                             // LDS broadcast b64
            a0 += qv.y * fast_tanh(qv.x + kv.x);
            a1 += qv.y * fast_tanh(qv.x + kv.y);
        }
        float2 r = make_float2(a0 + wbias, a1 + wbias);
        *(float2*)&logits[((size_t)(b * TQ_ + q0 + wave)) * TK_ + kh * KCH_ + 2 * lane] = r;
    }
}

// ------------------- 4) softmax: one wave per row --------------------------
__global__ __launch_bounds__(256) void softmax_kernel(float* __restrict__ attn,
                                                      __bf16* __restrict__ pbf) {
    const int t = threadIdx.x, wave = t >> 5, lane = t & 31;
    const size_t row = (size_t)blockIdx.x * 8 + wave;
    float* rp = attn + row * TK_;

    float4 v[4];
#pragma unroll
    for (int c = 0; c < 4; ++c) v[c] = *(const float4*)(rp + c * 128 + 4 * lane);

    float m = v[0].x;
#pragma unroll
    for (int c = 0; c < 4; ++c) {
        m = fmaxf(m, v[c].x); m = fmaxf(m, v[c].y);
        m = fmaxf(m, v[c].z); m = fmaxf(m, v[c].w);
    }
    for (int off = 16; off > 0; off >>= 1) m = fmaxf(m, __shfl_xor(m, off, 32));

    const float L2E = 1.4426950408889634f;
    float s = 0.0f;
#pragma unroll
    for (int c = 0; c < 4; ++c) {
        v[c].x = __builtin_amdgcn_exp2f((v[c].x - m) * L2E);
        v[c].y = __builtin_amdgcn_exp2f((v[c].y - m) * L2E);
        v[c].z = __builtin_amdgcn_exp2f((v[c].z - m) * L2E);
        v[c].w = __builtin_amdgcn_exp2f((v[c].w - m) * L2E);
        s += v[c].x + v[c].y + v[c].z + v[c].w;
    }
    for (int off = 16; off > 0; off >>= 1) s += __shfl_xor(s, off, 32);
    const float inv = 1.0f / s;

#pragma unroll
    for (int c = 0; c < 4; ++c) {
        float4 p;
        p.x = v[c].x * inv; p.y = v[c].y * inv; p.z = v[c].z * inv; p.w = v[c].w * inv;
        *(float4*)(rp + c * 128 + 4 * lane) = p;
        v4bf o;
        o[0] = (__bf16)p.x; o[1] = (__bf16)p.y; o[2] = (__bf16)p.z; o[3] = (__bf16)p.w;
        *(v4bf*)(pbf + row * TK_ + c * 128 + 4 * lane) = o;
    }
}

// ------------------- 5) out = attn @ value (bf16 WMMA) ---------------------
// Grid (4,4,8): per batch, 128x128 output tile/block; wave = 16 rows x 128 cols.
__global__ __launch_bounds__(256) void av_kernel(const __bf16* __restrict__ P,
                                                 const __bf16* __restrict__ V,
                                                 float* __restrict__ O) {
    __shared__ __bf16 VT[32][136];                         // value chunk [k][n]
    const int t = threadIdx.x, wave = t >> 5, lane = t & 31;
    const int b = blockIdx.z, mt = blockIdx.y, nt = blockIdx.x;
    const __bf16* Pb = P + (size_t)b * TQ_ * TK_;
    const __bf16* Vb = V + (size_t)b * TK_ * DD_;
    float*        Ob = O + (size_t)b * TQ_ * DD_;

    const int m0 = mt * 128 + wave * 16, n0 = nt * 128;
    const int mrow  = m0 + (lane & 15);
    const int kbase = (lane < 16) ? 0 : 8;
    const int hi8   = (lane < 16) ? 0 : 8;

    v8f acc[8];
#pragma unroll
    for (int i = 0; i < 8; ++i)
#pragma unroll
        for (int e = 0; e < 8; ++e) acc[i][e] = 0.0f;

    for (int kc = 0; kc < 16; ++kc) {
        const int k0 = kc * 32;
        __syncthreads();
        for (int idx = t; idx < 32 * 128; idx += 256) {    // coalesced bf16 stage
            int kk = idx >> 7, nn = idx & 127;
            VT[kk][nn] = Vb[(size_t)(k0 + kk) * DD_ + n0 + nn];
        }
        __syncthreads();

        v16bf afr;
        const __bf16* pr = Pb + (size_t)mrow * TK_ + k0 + kbase;
#pragma unroll
        for (int p = 0; p < 8; ++p) {
            int kp = (p < 4) ? 2 * p : 16 + 2 * (p - 4);
            v2bf pv = *(const v2bf*)(pr + kp);
            afr[2 * p]     = pv[0];
            afr[2 * p + 1] = pv[1];
        }
#pragma unroll
        for (int nb = 0; nb < 8; ++nb) {
            v16bf bfr;
#pragma unroll
            for (int s = 0; s < 16; ++s)
                bfr[s] = VT[kbase + ((s < 8) ? s : 8 + s)][nb * 16 + (lane & 15)];
            acc[nb] = __builtin_amdgcn_wmma_f32_16x16x32_bf16(
                false, afr, false, bfr, (short)0, acc[nb], false, false);
        }
    }
#pragma unroll
    for (int nb = 0; nb < 8; ++nb)
#pragma unroll
        for (int r = 0; r < 8; ++r)
            Ob[(size_t)(m0 + r + hi8) * DD_ + n0 + nb * 16 + (lane & 15)] = acc[nb][r];
}

// ---------------------------------------------------------------------------
extern "C" void kernel_launch(void* const* d_in, const int* in_sizes, int n_in,
                              void* d_out, int out_size, void* d_ws, size_t ws_size,
                              hipStream_t stream) {
    const float* query = (const float*)d_in[0];
    const float* key   = (const float*)d_in[1];
    const float* value = (const float*)d_in[2];
    const float* Wq    = (const float*)d_in[3];
    const float* Wk    = (const float*)d_in[4];
    const float* bias  = (const float*)d_in[5];
    const float* w_w   = (const float*)d_in[6];
    const float* w_b   = (const float*)d_in[7];

    float* out  = (float*)d_out;                           // [B,TQ,DV]
    float* attn = out + (size_t)B_ * TQ_ * DD_;            // [B,TQ,TK]

    // workspace layout (~12.6 MB)
    float*  q_ws = (float*)d_ws;                           // [B*TQ, A] fp32
    float*  k_ws = q_ws + (size_t)B_ * TQ_ * AD_;          // [B*TK, A] fp32
    __bf16* v_bf = (__bf16*)(k_ws + (size_t)B_ * TK_ * AD_);   // [B,TK,DV] bf16
    __bf16* p_bf = v_bf + (size_t)B_ * TK_ * DD_;              // [B,TQ,TK] bf16

    cvt_kernel<<<(B_ * TK_ * DD_) / (256 * 4), 256, 0, stream>>>(value, v_bf);
    proj_kernel<<<(B_ * TQ_) / 128, 256, 0, stream>>>(query, Wq, q_ws);
    proj_kernel<<<(B_ * TK_) / 128, 256, 0, stream>>>(key, Wk, k_ws);
    dim3 g2(TQ_ / QB_, B_);
    logits_kernel<<<g2, 256, 0, stream>>>(q_ws, k_ws, bias, w_w, w_b, attn);
    softmax_kernel<<<(B_ * TQ_) / 8, 256, 0, stream>>>(attn, p_bf);
    dim3 g3(DD_ / 128, TQ_ / 128, B_);
    av_kernel<<<g3, 256, 0, stream>>>(p_bf, v_bf, out);
}
